// SparseLogicLayer_90580860273064
// MI455X (gfx1250) — matrix operand order repaired and checked by reading, not compiled, not run on hardware
//
#include <hip/hip_runtime.h>

// Problem constants (match reference)
#define Bb 4
#define Nn 768
#define Cc 8

typedef __attribute__((ext_vector_type(16))) _Float16 v16h;
typedef __attribute__((ext_vector_type(8)))  float    v8f;
typedef __attribute__((ext_vector_type(4)))  float    v4f;

// Fast sigmoid: v_exp_f32 + v_rcp_f32 (no IEEE div expansion).
// ~1 ulp rcp error << f16 input quantization already present in the WMMA path.
__device__ __forceinline__ float sigmoidf(float x) {
    return __builtin_amdgcn_rcpf(1.0f + __expf(-x));
}

// ---------------------------------------------------------------------------
// Order-2 head: one wave handles (b, i, j0..j0+15) via a single
// V_WMMA_F32_16X16X32_F16:  D[9 ch][16 pos] = Wt[9][32] x F2[32][16] + bias.
// ---------------------------------------------------------------------------
__global__ __launch_bounds__(256)
void order2_wmma_kernel(const float* __restrict__ x1,
                        const float* __restrict__ x2,
                        const float* __restrict__ Wl2,
                        const float* __restrict__ bl2,
                        const float* __restrict__ Wa2,
                        const float* __restrict__ ba2,
                        float* __restrict__ o2,
                        float* __restrict__ a2)
{
    const int JT   = Nn / 16;                                   // 48 j-tiles per row
    const int wave = (blockIdx.x * blockDim.x + threadIdx.x) >> 5;
    const int lane = threadIdx.x & 31;
    const int n    = lane & 15;

    const int jt  = wave % JT;
    const int t1  = wave / JT;
    const int i   = t1 % Nn;
    const int b   = t1 / Nn;
    const int jj  = jt * 16 + n;

    // ---- A fragment: transposed weights (rows = output channels 0..8) ----
    // lane<16 : row M=lane, K = 0..15; lane>=16 : K = 16..31
    v16h afrag;
    #pragma unroll
    for (int h = 0; h < 16; ++h) {
        int kk = h;
        if (lane >= 16) kk += 16;
        float w = 0.0f;
        if (n < 8)       w = Wl2[kk * 8 + n];   // logic channel n
        else if (n == 8) w = Wa2[kk];           // alpha channel
        afrag[h] = (_Float16)w;
    }

    // ---- B fragment: column n = position (i, jj), 32 features ----
    // lane<16 : K0..15  = [x1[b,i],  x2[b,i,jj]]
    // lane>=16: K16..31 = [x1[b,jj], x2[b,jj,i]]
    const float* p_x1;
    const float* p_x2;
    if (lane < 16) {
        p_x1 = x1 + ((size_t)b * Nn + i) * Cc;
        p_x2 = x2 + (((size_t)b * Nn + i) * Nn + jj) * Cc;   // coalesced slab
    } else {
        p_x1 = x1 + ((size_t)b * Nn + jj) * Cc;
        p_x2 = x2 + (((size_t)b * Nn + jj) * Nn + i) * Cc;   // transposed; L2-resident
    }
    v4f xa0 = ((const v4f*)p_x1)[0];
    v4f xa1 = ((const v4f*)p_x1)[1];
    v4f xb0 = ((const v4f*)p_x2)[0];
    v4f xb1 = ((const v4f*)p_x2)[1];
    float xv[16] = {xa0[0], xa0[1], xa0[2], xa0[3], xa1[0], xa1[1], xa1[2], xa1[3],
                    xb0[0], xb0[1], xb0[2], xb0[3], xb1[0], xb1[1], xb1[2], xb1[3]};
    v16h bfrag;
    #pragma unroll
    for (int h = 0; h < 16; ++h) bfrag[h] = (_Float16)xv[h];

    // ---- C accumulator pre-loaded with biases ----
    // lane<16: rows 0..7 -> bl2[r]; lane>=16: rows 8..15 -> r==0 is ba2
    v8f cfrag;
    #pragma unroll
    for (int r = 0; r < 8; ++r)
        cfrag[r] = (lane < 16) ? bl2[r] : ((r == 0) ? ba2[0] : 0.0f);

    v8f dfrag = __builtin_amdgcn_wmma_f32_16x16x32_f16(
        /*neg_a=*/false, afrag, /*neg_b=*/false, bfrag,
        /*c_mod=*/(short)0, cfrag, /*reuse_a=*/false, /*reuse_b=*/false);

    // alpha logit for position n lives in lane n+16, component 0
    float d0 = dfrag[0];
    float alphaLogit = __shfl(d0, lane | 16, 32);

    if (lane < 16) {
        float a = sigmoidf(alphaLogit);
        size_t pos = ((size_t)b * Nn + i) * Nn + jj;
        v4f lo, hi;
        lo[0] = sigmoidf(dfrag[0]) * a;  lo[1] = sigmoidf(dfrag[1]) * a;
        lo[2] = sigmoidf(dfrag[2]) * a;  lo[3] = sigmoidf(dfrag[3]) * a;
        hi[0] = sigmoidf(dfrag[4]) * a;  hi[1] = sigmoidf(dfrag[5]) * a;
        hi[2] = sigmoidf(dfrag[6]) * a;  hi[3] = sigmoidf(dfrag[7]) * a;
        v4f* op = (v4f*)(o2 + pos * Cc);
        __builtin_nontemporal_store(lo, op);        // stream: keep x2 in L2
        __builtin_nontemporal_store(hi, op + 1);
        __builtin_nontemporal_store(a, a2 + pos);
    }
}

// ---------------------------------------------------------------------------
// Order-1 head: one block per (b, i); masked max/min reduce of x2 over j,
// then 9-output scalar GEMM (K=32).
// ---------------------------------------------------------------------------
__global__ __launch_bounds__(256)
void order1_kernel(const float* __restrict__ x0, const float* __restrict__ x1,
                   const float* __restrict__ x2,
                   const float* __restrict__ Wl1, const float* __restrict__ bl1,
                   const float* __restrict__ Wa1, const float* __restrict__ ba1,
                   float* __restrict__ o1, float* __restrict__ a1)
{
    __shared__ float smx[256], smn[256], f1s[32], sAlpha;
    const int t = threadIdx.x;
    const int b = blockIdx.x / Nn;
    const int i = blockIdx.x % Nn;
    const int c = t & 7;
    const int g = t >> 3;                          // 32 j-groups

    float mx = -1e30f, mn = 1e30f;
    const float* row = x2 + ((size_t)b * Nn + i) * Nn * Cc;
    for (int j = g; j < Nn; j += 32) {
        float v  = row[(size_t)j * Cc + c];
        float vx = (j == i) ? 0.0f : v;            // matches where(mask, x2, 0.0)
        float vn = (j == i) ? 1.0f : v;            // matches where(mask, x2, 1.0)
        mx = fmaxf(mx, vx);
        mn = fminf(mn, vn);
    }
    smx[t] = mx; smn[t] = mn;
    __syncthreads();
    for (int s = 128; s >= 8; s >>= 1) {
        if (t < s) {
            smx[t] = fmaxf(smx[t], smx[t + s]);
            smn[t] = fminf(smn[t], smn[t + s]);
        }
        __syncthreads();
    }
    if (t < 8) {                                   // f1 = [x0, x1, interleave(ex,fa)]
        f1s[t]          = x0[b * Cc + t];
        f1s[8 + t]      = x1[((size_t)b * Nn + i) * Cc + t];
        f1s[16 + 2 * t]     = smx[t];
        f1s[16 + 2 * t + 1] = smn[t];
    }
    __syncthreads();
    if (t == 8) {
        float acc = ba1[0];
        #pragma unroll
        for (int k = 0; k < 32; ++k) acc += f1s[k] * Wa1[k];
        sAlpha = sigmoidf(acc);
    }
    __syncthreads();
    if (t < 8) {
        float acc = bl1[t];
        #pragma unroll
        for (int k = 0; k < 32; ++k) acc += f1s[k] * Wl1[k * 8 + t];
        o1[((size_t)b * Nn + i) * Cc + t] = sigmoidf(acc) * sAlpha;
    }
    if (t == 8) a1[(size_t)b * Nn + i] = sAlpha;
}

// ---------------------------------------------------------------------------
// Order-0 head: one block per b; reduce x1 over N, then 9-output GEMM (K=24).
// ---------------------------------------------------------------------------
__global__ __launch_bounds__(256)
void order0_kernel(const float* __restrict__ x0, const float* __restrict__ x1,
                   const float* __restrict__ Wl0, const float* __restrict__ bl0,
                   const float* __restrict__ Wa0, const float* __restrict__ ba0,
                   float* __restrict__ o0, float* __restrict__ a0)
{
    __shared__ float smx[256], smn[256], f0s[24], sAlpha;
    const int t = threadIdx.x;
    const int b = blockIdx.x;
    const int c = t & 7;
    const int g = t >> 3;

    float mx = -1e30f, mn = 1e30f;
    const float* base = x1 + (size_t)b * Nn * Cc;
    for (int j = g; j < Nn; j += 32) {
        float v = base[(size_t)j * Cc + c];
        mx = fmaxf(mx, v);
        mn = fminf(mn, v);
    }
    smx[t] = mx; smn[t] = mn;
    __syncthreads();
    for (int s = 128; s >= 8; s >>= 1) {
        if (t < s) {
            smx[t] = fmaxf(smx[t], smx[t + s]);
            smn[t] = fminf(smn[t], smn[t + s]);
        }
        __syncthreads();
    }
    if (t < 8) {                                   // f0 = [x0, interleave(ex,fa)]
        f0s[t]             = x0[b * Cc + t];
        f0s[8 + 2 * t]     = smx[t];
        f0s[8 + 2 * t + 1] = smn[t];
    }
    __syncthreads();
    if (t == 8) {
        float acc = ba0[0];
        #pragma unroll
        for (int k = 0; k < 24; ++k) acc += f0s[k] * Wa0[k];
        sAlpha = sigmoidf(acc);
    }
    __syncthreads();
    if (t < 8) {
        float acc = bl0[t];
        #pragma unroll
        for (int k = 0; k < 24; ++k) acc += f0s[k] * Wl0[k * 8 + t];
        o0[b * Cc + t] = sigmoidf(acc) * sAlpha;
    }
    if (t == 8) a0[b] = sAlpha;
}

extern "C" void kernel_launch(void* const* d_in, const int* in_sizes, int n_in,
                              void* d_out, int out_size, void* d_ws, size_t ws_size,
                              hipStream_t stream) {
    const float* x0  = (const float*)d_in[0];
    const float* x1  = (const float*)d_in[1];
    const float* x2  = (const float*)d_in[2];
    const float* Wl0 = (const float*)d_in[3];
    const float* bl0 = (const float*)d_in[4];
    const float* Wa0 = (const float*)d_in[5];
    const float* ba0 = (const float*)d_in[6];
    const float* Wl1 = (const float*)d_in[7];
    const float* bl1 = (const float*)d_in[8];
    const float* Wa1 = (const float*)d_in[9];
    const float* ba1 = (const float*)d_in[10];
    const float* Wl2 = (const float*)d_in[11];
    const float* bl2 = (const float*)d_in[12];
    const float* Wa2 = (const float*)d_in[13];
    const float* ba2 = (const float*)d_in[14];

    // Flat output layout: o0 | o1 | o2 | a0 | a1 | a2
    constexpr size_t n_o0 = (size_t)Bb * Cc;                 // 32
    constexpr size_t n_o1 = (size_t)Bb * Nn * Cc;            // 24576
    constexpr size_t n_o2 = (size_t)Bb * Nn * Nn * Cc;       // 18874368
    constexpr size_t n_a0 = (size_t)Bb;                      // 4
    constexpr size_t n_a1 = (size_t)Bb * Nn;                 // 3072
    float* out = (float*)d_out;
    float* o0 = out;
    float* o1 = o0 + n_o0;
    float* o2 = o1 + n_o1;
    float* a0 = o2 + n_o2;
    float* a1 = a0 + n_a0;
    float* a2 = a1 + n_a1;

    // Order 2: 4*768*48 = 147456 waves, 8 waves/block -> 18432 blocks
    const int totalWaves = Bb * Nn * (Nn / 16);
    order2_wmma_kernel<<<totalWaves / 8, 256, 0, stream>>>(
        x1, x2, Wl2, bl2, Wa2, ba2, o2, a2);
    order1_kernel<<<Bb * Nn, 256, 0, stream>>>(
        x0, x1, x2, Wl1, bl1, Wa1, ba1, o1, a1);
    order0_kernel<<<Bb, 256, 0, stream>>>(
        x0, x1, Wl0, bl0, Wa0, ba0, o0, a0);
}
